// OutlookAttention_26800595927449
// MI455X (gfx1250) — matrix operand order repaired and checked by reading, not compiled, not run on hardware
//
#include <hip/hip_runtime.h>
#include <hip/hip_bf16.h>

// ---- problem constants ----
#define BB     8
#define HH     56
#define WW     56
#define CC     192
#define HEADS  6
#define KWIN   3
#define KKW    9            // K*K
#define HWPIX  (HH*WW)      // 3136
#define NPIX   (BB*HWPIX)   // 25088
#define AC     (KKW*KKW*HEADS)  // 486
#define AC_PAD 512          // 8 * 64  (so N block-tile of 64 needs no load guards)
#define SCALE  0.17677669529663687f  // 32^-0.5

typedef __attribute__((ext_vector_type(16))) __bf16 bf16x16;
typedef __attribute__((ext_vector_type(8)))  __bf16 bf16x8;
typedef __attribute__((ext_vector_type(8)))  float  f32x8;

__device__ __forceinline__ unsigned short f2bf(float f) {
    unsigned int u = __float_as_uint(f);
    unsigned int r = u + 0x7FFFu + ((u >> 16) & 1u);   // round-to-nearest-even
    return (unsigned short)(r >> 16);
}

// ---- fp32 -> bf16 raw convert ----
__global__ void k_f32_to_bf16(const float* __restrict__ src,
                              unsigned short* __restrict__ dst, int n) {
    int i = blockIdx.x * blockDim.x + threadIdx.x;
    if (i < n) dst[i] = f2bf(src[i]);
}

// ---- weight [192][N] row-major -> transposed bf16 [Npad][192], zero pad rows ----
__global__ void k_weight_transpose_bf16(const float* __restrict__ w,
                                        unsigned short* __restrict__ wt,
                                        int N, int Npad) {
    int i = blockIdx.x * blockDim.x + threadIdx.x;
    int total = Npad * CC;
    if (i >= total) return;
    int n = i / CC, k = i % CC;
    float v = (n < N) ? w[(size_t)k * N + n] : 0.f;
    wt[(size_t)n * CC + k] = f2bf(v);
}

// A fragment (16x32 bf16): lane ll holds row; element i maps to
// k = (i/8)*16 + lh*8 + (i%8)  -> two contiguous 8-element runs.
__device__ __forceinline__ bf16x16 load_afrag(const unsigned short* arow, int k0, int lh) {
    bf16x8 lo = *(const bf16x8*)(arow + k0 + lh * 8);
    bf16x8 hi = *(const bf16x8*)(arow + k0 + 16 + lh * 8);
    bf16x16 av;
    #pragma unroll
    for (int i = 0; i < 8; ++i) { av[i] = lo[i]; av[i + 8] = hi[i]; }
    return av;
}

// ---- bf16 WMMA GEMM: C[M][N] = A[M][192] * B[192][N] (+bias), fp32 accumulate ----
// A: bf16 row-major [M][192]; BT: bf16 [Npad][192] (B transposed). K fixed = 192.
// 128 threads = 4 waves in 2x2; each wave computes a 32x32 tile (2 A-frags x
// 2 B-frags -> 4 WMMAs per K-step) for 2x operand reuse vs a 16x16 tile.
// Block tile = 64(M) x 64(N).
__global__ __launch_bounds__(128)
void k_gemm_bf16_wmma(const unsigned short* __restrict__ A,
                      const unsigned short* __restrict__ BT,
                      const float* __restrict__ bias,
                      float* __restrict__ C, int N) {
    const int lane = threadIdx.x & 31;
    const int wave = threadIdx.x >> 5;
    const int m0 = blockIdx.x * 64 + (wave >> 1) * 32;
    const int n0 = blockIdx.y * 64 + (wave & 1) * 32;
    const int lh = lane >> 4;   // half-wave select (K-half)
    const int ll = lane & 15;   // row (A) / col (B) within 16-tile

    const unsigned short* arow0 = A  + (size_t)(m0 + ll) * CC;
    const unsigned short* arow1 = arow0 + (size_t)16 * CC;
    const unsigned short* brow0 = BT + (size_t)(n0 + ll) * CC;
    const unsigned short* brow1 = brow0 + (size_t)16 * CC;
    __builtin_prefetch(arow0, 0, 3);   // global_prefetch_b8
    __builtin_prefetch(brow0, 0, 3);

    f32x8 acc00 = {}, acc01 = {}, acc10 = {}, acc11 = {};
    #pragma unroll
    for (int k0 = 0; k0 < CC; k0 += 32) {
        bf16x16 a0 = load_afrag(arow0, k0, lh);
        bf16x16 a1 = load_afrag(arow1, k0, lh);
        // B fragment (32x16 bf16): lane holds column, element i -> k = (i/8)*16 + lh*8 + (i%8).
        bf16x16 b0 = load_afrag(brow0, k0, lh);
        bf16x16 b1 = load_afrag(brow1, k0, lh);
        acc00 = __builtin_amdgcn_wmma_f32_16x16x32_bf16(false, a0, false, b0, (short)0, acc00, false, false);
        acc01 = __builtin_amdgcn_wmma_f32_16x16x32_bf16(false, a0, false, b1, (short)0, acc01, false, false);
        acc10 = __builtin_amdgcn_wmma_f32_16x16x32_bf16(false, a1, false, b0, (short)0, acc10, false, false);
        acc11 = __builtin_amdgcn_wmma_f32_16x16x32_bf16(false, a1, false, b1, (short)0, acc11, false, false);
    }

    // C layout per 16x16 tile: VGPR r -> M = r + 8*lh, N = ll
    auto store_tile = [&](const f32x8& acc, int mt, int nt) {
        int col = nt + ll;
        if (col < N) {
            float bv = bias ? bias[col] : 0.f;
            #pragma unroll
            for (int r = 0; r < 8; ++r) {
                int row = mt + r + 8 * lh;
                C[(size_t)row * N + col] = acc[r] + bv;
            }
        }
    };
    store_tile(acc00, m0,      n0);
    store_tile(acc01, m0,      n0 + 16);
    store_tile(acc10, m0 + 16, n0);
    store_tile(acc11, m0 + 16, n0 + 16);
}

// ---- softmax over last dim (9), with scale, in place ----
__global__ void k_softmax9(float* __restrict__ a, int rows) {
    int i = blockIdx.x * blockDim.x + threadIdx.x;
    if (i >= rows) return;
    int p = i / (HEADS * KKW);
    int r = i % (HEADS * KKW);
    int h = r / KKW, k = r % KKW;
    float* row = a + (size_t)p * AC + h * (KKW * KKW) + k * KKW;
    float v[9], m = -1e30f;
    #pragma unroll
    for (int l = 0; l < 9; ++l) { v[l] = row[l] * SCALE; m = fmaxf(m, v[l]); }
    float s = 0.f;
    #pragma unroll
    for (int l = 0; l < 9; ++l) { v[l] = __expf(v[l] - m); s += v[l]; }
    float inv = 1.f / s;
    #pragma unroll
    for (int l = 0; l < 9; ++l) row[l] = v[l] * inv;
}

// ---- fused attention-apply + fold (gather form), bf16 output for final GEMM ----
// out[q,c] = sum_{(i,j)} sum_l a[b,h,p=q-off(i,j),(i,j),l] * v[b, q+off(l)-off(i,j), c]
__global__ __launch_bounds__(192)
void k_attn_fold(const float* __restrict__ a,        // [NPIX][486] softmaxed
                 const float* __restrict__ v,        // [NPIX][192]
                 unsigned short* __restrict__ ofold) // [NPIX][192] bf16
{
    __shared__ float v_lds[25 * CC];   // 5x5 neighborhood of v
    __shared__ float a_lds[AC];        // 9 source pixels x 6 heads x 9
    const int t = threadIdx.x;         // channel
    const int pix = blockIdx.x;
    const int b = pix / HWPIX;
    const int q = pix % HWPIX;
    const int Y = q / WW, X = q % WW;

    // v neighborhood (unfold zero-padding handled here)
    for (int n = 0; n < 25; ++n) {
        int vy = Y + n / 5 - 2, vx = X + n % 5 - 2;
        float val = 0.f;
        if (vy >= 0 && vy < HH && vx >= 0 && vx < WW)
            val = v[(size_t)(b * HWPIX + vy * WW + vx) * CC + t];
        v_lds[n * CC + t] = val;
    }
    // a slices: for source (i,j) we need kernel-index k=(i,j) row of length 9 per head
    for (int e = t; e < AC; e += 192) {
        int si = e / 54;               // source / kernel position (i*3+j)
        int r  = e % 54;
        int h  = r / 9, l = r % 9;
        int py = Y - (si / 3 - 1), px = X - (si % 3 - 1);
        float val = 0.f;               // fold validity: OOB source contributes 0
        if (py >= 0 && py < HH && px >= 0 && px < WW)
            val = a[(size_t)(b * HWPIX + py * WW + px) * AC + h * 81 + si * 9 + l];
        a_lds[e] = val;
    }
    __syncthreads();

    const int h9 = (t >> 5) * 9;       // head * 9
    float acc = 0.f;
    #pragma unroll
    for (int si = 0; si < 9; ++si) {
        const int iy = si / 3, ix = si % 3;
        #pragma unroll
        for (int l = 0; l < 9; ++l) {
            const int ly = l / 3, lx = l % 3;
            const int vidx = (ly - iy + 2) * 5 + (lx - ix + 2);
            acc = fmaf(a_lds[si * 54 + h9 + l], v_lds[vidx * CC + t], acc);
        }
    }
    ofold[(size_t)pix * CC + t] = f2bf(acc);
}

extern "C" void kernel_launch(void* const* d_in, const int* in_sizes, int n_in,
                              void* d_out, int out_size, void* d_ws, size_t ws_size,
                              hipStream_t stream) {
    const float* x  = (const float*)d_in[0];
    const float* Wv = (const float*)d_in[1];
    const float* Wa = (const float*)d_in[2];
    const float* ba = (const float*)d_in[3];
    const float* Wp = (const float*)d_in[4];
    const float* bp = (const float*)d_in[5];
    float* out = (float*)d_out;
    (void)in_sizes; (void)n_in; (void)out_size; (void)ws_size;

    char* ws = (char*)d_ws;
    size_t off = 0;
    auto alloc = [&](size_t bytes) -> char* {
        char* p = ws + off;
        off = (off + bytes + 255) & ~(size_t)255;
        return p;
    };
    unsigned short* xbf   = (unsigned short*)alloc((size_t)NPIX * CC * 2);
    unsigned short* WvT   = (unsigned short*)alloc((size_t)CC * CC * 2);
    unsigned short* WaT   = (unsigned short*)alloc((size_t)AC_PAD * CC * 2);
    unsigned short* WpT   = (unsigned short*)alloc((size_t)CC * CC * 2);
    float*          vbuf  = (float*)alloc((size_t)NPIX * CC * 4);
    float*          abuf  = (float*)alloc((size_t)NPIX * AC * 4);
    unsigned short* ofold = (unsigned short*)alloc((size_t)NPIX * CC * 2);

    // precision/layout prep: bf16 activations, transposed bf16 weights
    {
        int n = NPIX * CC;
        k_f32_to_bf16<<<(n + 255) / 256, 256, 0, stream>>>(x, xbf, n);
    }
    k_weight_transpose_bf16<<<(CC * CC + 255) / 256, 256, 0, stream>>>(Wv, WvT, CC, CC);
    k_weight_transpose_bf16<<<(AC_PAD * CC + 255) / 256, 256, 0, stream>>>(Wa, WaT, AC, AC_PAD);
    k_weight_transpose_bf16<<<(CC * CC + 255) / 256, 256, 0, stream>>>(Wp, WpT, CC, CC);

    // v = x @ Wv            (WMMA)
    dim3 g1(NPIX / 64, CC / 64);        // 392 x 3
    k_gemm_bf16_wmma<<<g1, 128, 0, stream>>>(xbf, WvT, nullptr, vbuf, CC);
    // a = x @ Wa + ba       (WMMA, N=486 guarded; loads padded to 512)
    dim3 g2(NPIX / 64, AC_PAD / 64);    // 392 x 8
    k_gemm_bf16_wmma<<<g2, 128, 0, stream>>>(xbf, WaT, ba, abuf, AC);

    // softmax(a * scale) over window dim
    int rows = NPIX * HEADS * KKW;
    k_softmax9<<<(rows + 255) / 256, 256, 0, stream>>>(abuf, rows);

    // fused attention-apply + col2im fold -> bf16
    k_attn_fold<<<NPIX, 192, 0, stream>>>(abuf, vbuf, ofold);

    // out = fold(o) @ Wp + bp   (WMMA)
    k_gemm_bf16_wmma<<<g1, 128, 0, stream>>>(ofold, WpT, bp, out, CC);
}
